// DenseLocal_11416023073154
// MI455X (gfx1250) — matrix operand order repaired and checked by reading, not compiled, not run on hardware
//
#include <hip/hip_runtime.h>

// bf16 x3 split-precision fp32 batched GEMM + softplus for MI455X (gfx1250, wave32).
// LDS ping-pong buffers hold pre-split bf16 hi/lo planes; compute phase is pure
// ds_load_b128 + v_wmma_f32_16x16x32_bf16, one barrier per K-chunk.
typedef __attribute__((ext_vector_type(16))) __bf16 v16bf;
typedef __attribute__((ext_vector_type(8)))  __bf16 v8bf;
typedef __attribute__((ext_vector_type(4)))  __bf16 v4bf;
typedef __attribute__((ext_vector_type(8)))  float  v8f;

#define BM 128
#define BN 128
#define BK 32
#define LSTR 40   // LDS row stride in bf16 elements (80 B): 16B-aligned frag reads, good bank sweep

static __device__ __forceinline__ float f4_get(const float4& v, int c) {
  return c == 0 ? v.x : c == 1 ? v.y : c == 2 ? v.z : v.w;
}

// A-fragment (16x32 bf16): 16 contiguous halves per lane
static __device__ __forceinline__ v16bf load_frag_a(const __bf16* p) {
  v8bf a = *(const v8bf*)(p);
  v8bf b = *(const v8bf*)(p + 8);
  return __builtin_shufflevector(a, b, 0,1,2,3,4,5,6,7,8,9,10,11,12,13,14,15);
}
// B-fragment (32x16 bf16): 16 contiguous halves per lane
static __device__ __forceinline__ v16bf load_frag_b(const __bf16* p) {
  v8bf a = *(const v8bf*)(p);
  v8bf b = *(const v8bf*)(p + 8);
  return __builtin_shufflevector(a, b, 0,1,2,3,4,5,6,7,8,9,10,11,12,13,14,15);
}

__global__ __launch_bounds__(256)
void dense_local_wmma_bf16x3(const float* __restrict__ x,
                             const float* __restrict__ w,
                             float* __restrict__ out) {
  // ping-pong: 2 buffers x 4 planes x 10 KB = 80 KB (WGP has 320 KB)
  __shared__ __bf16 Xhi[2][BM * LSTR];
  __shared__ __bf16 Xlo[2][BM * LSTR];
  __shared__ __bf16 Whi[2][BN * LSTR];   // transposed: [o][i]
  __shared__ __bf16 Wlo[2][BN * LSTR];

  const int tid  = threadIdx.x;
  const int lane = tid & 31;
  const int wave = tid >> 5;
  const int wm   = wave & 3;    // 4 waves along M: 32-row strip each
  const int wn   = wave >> 2;   // 2 waves along N: 64-col strip each

  const int n0 = blockIdx.x * BN;
  const int m0 = blockIdx.y * BM;
  const int kz = blockIdx.z;

  const int l16 = lane & 15;
  const int lhi = lane >> 4;

  v8f acc[2][4];
  #pragma unroll
  for (int i = 0; i < 2; ++i)
    #pragma unroll
    for (int j = 0; j < 4; ++j)
      acc[i][j] = (v8f){0.f,0.f,0.f,0.f,0.f,0.f,0.f,0.f};

  // ---- global->register double buffering ----
  // X: 128 rows x 32 f32 = 1024 float4; thread j-loop: row = lin>>3, seg = lin&7
  // W: each thread owns a 4x4 f32 block: rows i = kk + rg*4 + j, cols o = n0 + cg*4..+3
  const int rg = tid >> 5;      // 0..7  (i-group)
  const int cg = tid & 31;      // 0..31 (o-group)
  float4 xr[4], wr[4];
  #pragma unroll
  for (int j = 0; j < 4; ++j) {
    const int lin = tid + j * 256;
    const int xrow = lin >> 3, xseg = lin & 7;
    xr[j] = *(const float4*)(x + ((size_t)(m0 + xrow) * 128 + kz) * 1024 + xseg * 4);
    wr[j] = *(const float4*)(w + ((size_t)kz * 1024 + rg * 4 + j) * 1024 + n0 + cg * 4);
  }

  int buf = 0;
  for (int kk = 0; kk < 1024; kk += BK) {
    // ---- split f32 -> bf16 hi/lo once per block, store planes to LDS[buf] ----
    #pragma unroll
    for (int j = 0; j < 4; ++j) {
      const int lin = tid + j * 256;
      const int xrow = lin >> 3, xseg = lin & 7;
      const float4 v = xr[j];
      const __bf16 h0 = (__bf16)v.x, h1 = (__bf16)v.y, h2 = (__bf16)v.z, h3 = (__bf16)v.w;
      *(v4bf*)(&Xhi[buf][xrow * LSTR + xseg * 4]) = (v4bf){h0, h1, h2, h3};
      *(v4bf*)(&Xlo[buf][xrow * LSTR + xseg * 4]) =
          (v4bf){(__bf16)(v.x - (float)h0), (__bf16)(v.y - (float)h1),
                 (__bf16)(v.z - (float)h2), (__bf16)(v.w - (float)h3)};
    }
    #pragma unroll
    for (int c = 0; c < 4; ++c) {      // register-blocked 4x4 transpose of W
      const float e0 = f4_get(wr[0], c), e1 = f4_get(wr[1], c),
                  e2 = f4_get(wr[2], c), e3 = f4_get(wr[3], c);
      const __bf16 h0 = (__bf16)e0, h1 = (__bf16)e1, h2 = (__bf16)e2, h3 = (__bf16)e3;
      const int orow = cg * 4 + c;
      *(v4bf*)(&Whi[buf][orow * LSTR + rg * 4]) = (v4bf){h0, h1, h2, h3};
      *(v4bf*)(&Wlo[buf][orow * LSTR + rg * 4]) =
          (v4bf){(__bf16)(e0 - (float)h0), (__bf16)(e1 - (float)h1),
                 (__bf16)(e2 - (float)h2), (__bf16)(e3 - (float)h3)};
    }
    __syncthreads();   // single barrier per chunk (ping-pong makes the WAR safe)

    // ---- prefetch next K-chunk during compute ----
    if (kk + BK < 1024) {
      const int kn = kk + BK;
      #pragma unroll
      for (int j = 0; j < 4; ++j) {
        const int lin = tid + j * 256;
        const int xrow = lin >> 3, xseg = lin & 7;
        xr[j] = *(const float4*)(x + ((size_t)(m0 + xrow) * 128 + kz) * 1024 + kn + xseg * 4);
        wr[j] = *(const float4*)(w + ((size_t)kz * 1024 + kn + rg * 4 + j) * 1024 + n0 + cg * 4);
      }
    }

    // ---- pure ds_load + WMMA compute phase ----
    v16bf Ahi[2], Alo[2], Bhi[4], Blo[4];
    #pragma unroll
    for (int mt = 0; mt < 2; ++mt) {
      const int off = (wm * 32 + mt * 16 + l16) * LSTR + lhi * 16;
      Ahi[mt] = load_frag_a(&Xhi[buf][off]);
      Alo[mt] = load_frag_a(&Xlo[buf][off]);
    }
    #pragma unroll
    for (int nt = 0; nt < 4; ++nt) {
      const int off = (wn * 64 + nt * 16 + l16) * LSTR + lhi * 16;
      Bhi[nt] = load_frag_b(&Whi[buf][off]);
      Blo[nt] = load_frag_b(&Wlo[buf][off]);
    }

    // term-major order: 8 independent accumulators between any same-acc reuse
    #pragma unroll
    for (int mt = 0; mt < 2; ++mt)
      #pragma unroll
      for (int nt = 0; nt < 4; ++nt)
        acc[mt][nt] = __builtin_amdgcn_wmma_f32_16x16x32_bf16(
            false, Ahi[mt], false, Bhi[nt], (short)0, acc[mt][nt], false, false);
    #pragma unroll
    for (int mt = 0; mt < 2; ++mt)
      #pragma unroll
      for (int nt = 0; nt < 4; ++nt)
        acc[mt][nt] = __builtin_amdgcn_wmma_f32_16x16x32_bf16(
            false, Ahi[mt], false, Blo[nt], (short)0, acc[mt][nt], false, false);
    #pragma unroll
    for (int mt = 0; mt < 2; ++mt)
      #pragma unroll
      for (int nt = 0; nt < 4; ++nt)
        acc[mt][nt] = __builtin_amdgcn_wmma_f32_16x16x32_bf16(
            false, Alo[mt], false, Bhi[nt], (short)0, acc[mt][nt], false, false);

    buf ^= 1;
  }

  // ---- epilogue: cheap stable softplus + store ----
  #pragma unroll
  for (int mt = 0; mt < 2; ++mt) {
    #pragma unroll
    for (int nt = 0; nt < 4; ++nt) {
      const int o = n0 + wn * 64 + nt * 16 + l16;
      #pragma unroll
      for (int v = 0; v < 8; ++v) {
        const int row = m0 + wm * 32 + mt * 16 + v + lhi * 8;
        const float s  = acc[mt][nt][v];
        const float sp = fmaxf(s, 0.0f) + __logf(1.0f + __expf(-fabsf(s)));
        out[((size_t)row * 128 + kz) * 1024 + o] = sp;
      }
    }
  }
}

extern "C" void kernel_launch(void* const* d_in, const int* in_sizes, int n_in,
                              void* d_out, int out_size, void* d_ws, size_t ws_size,
                              hipStream_t stream) {
  const float* x = (const float*)d_in[0];   // [512, 128, 1024] f32
  const float* w = (const float*)d_in[1];   // [128, 1024, 1024] f32
  float* out = (float*)d_out;               // [512, 128, 1024] f32

  dim3 grid(1024 / BN, 512 / BM, 128);      // (8, 4, 128)
  dense_local_wmma_bf16x3<<<grid, 256, 0, stream>>>(x, w, out);
}
